// AttentionBlock_49675591746081
// MI455X (gfx1250) — matrix operand order repaired and checked by reading, not compiled, not run on hardware
//
#include <hip/hip_runtime.h>
#include <hip/hip_bf16.h>

typedef __bf16 bf16_t;
typedef __attribute__((ext_vector_type(2)))  __bf16 bf16x2;
typedef __attribute__((ext_vector_type(4)))  __bf16 bf16x4;
typedef __attribute__((ext_vector_type(16))) __bf16 v16bf;
typedef __attribute__((ext_vector_type(8)))  float  v8f;

#define T_TOK 4096
#define C_CH  512
#define NH    8
#define CH    64
#define NG    32
#define GW    16
#define SCALE_QK 0.35355339059327373f   // 1/sqrt(sqrt(64))

// A-operand K index, contiguous order k = 16*hf + 2*i (matches B operands)
__device__ __forceinline__ int a_kidx(int i, int hf) {
    return (hf << 4) + (i << 1);
}

// ---------------------------------------------------------------------------
// Kernel 1: GroupNorm stats -> per-channel affine coeffs  xn = x*a[c] + b[c]
// ---------------------------------------------------------------------------
__global__ __launch_bounds__(256)
void gn_stats_kernel(const float* __restrict__ x,
                     const float* __restrict__ gs,
                     const float* __restrict__ gb,
                     float* __restrict__ acoef,
                     float* __restrict__ bcoef) {
    __shared__ float s_sum[256];
    __shared__ float s_sq[256];
    const int g = blockIdx.x;
    const int tid = threadIdx.x;
    float s = 0.f, ss = 0.f;
    for (int e = tid; e < T_TOK * GW; e += 256) {
        int t = e >> 4;
        int c = (g << 4) + (e & 15);
        float v = x[t * C_CH + c];
        s += v; ss += v * v;
    }
    s_sum[tid] = s; s_sq[tid] = ss;
    __syncthreads();
    for (int off = 128; off > 0; off >>= 1) {
        if (tid < off) { s_sum[tid] += s_sum[tid + off]; s_sq[tid] += s_sq[tid + off]; }
        __syncthreads();
    }
    if (tid < GW) {
        const float inv_n = 1.f / (float)(T_TOK * GW);
        float mean = s_sum[0] * inv_n;
        float var  = s_sq[0] * inv_n - mean * mean;
        float rstd = rsqrtf(var + 1e-6f);
        int c = (g << 4) + tid;
        float a = rstd * gs[c];
        acoef[c] = a;
        bcoef[c] = gb[c] - mean * a;
    }
}

// ---------------------------------------------------------------------------
// Kernel 2: normalized activations as bf16 (vectorized x4)
// ---------------------------------------------------------------------------
__global__ __launch_bounds__(256)
void xn_norm_kernel(const float* __restrict__ x,
                    const float* __restrict__ acoef,
                    const float* __restrict__ bcoef,
                    bf16_t* __restrict__ xn) {
    int i4 = blockIdx.x * 256 + threadIdx.x;   // 0 .. T*C/4 - 1
    int base = i4 * 4;
    int c = base & (C_CH - 1);
    float4 xv = *(const float4*)&x[base];
    float4 av = *(const float4*)&acoef[c];
    float4 bv = *(const float4*)&bcoef[c];
    bf16x4 r;
    r[0] = (bf16_t)(xv.x * av.x + bv.x);
    r[1] = (bf16_t)(xv.y * av.y + bv.y);
    r[2] = (bf16_t)(xv.z * av.z + bv.z);
    r[3] = (bf16_t)(xv.w * av.w + bv.w);
    *(bf16x4*)&xn[base] = r;
}

// ---------------------------------------------------------------------------
// Kernel 3: pack W: wt[j][c_out][c_in] = w_qkv[c_in][3*c_out + j] (*scale q,k)
// ---------------------------------------------------------------------------
__global__ __launch_bounds__(256)
void pack_w_kernel(const float* __restrict__ w_qkv, bf16_t* __restrict__ wt) {
    int idx = blockIdx.x * 256 + threadIdx.x;
    int j   = idx >> 18;
    int rem = idx & 262143;
    int co  = rem >> 9;
    int ci  = rem & 511;
    float v = w_qkv[ci * (3 * C_CH) + 3 * co + j];
    if (j < 2) v *= SCALE_QK;
    wt[idx] = (bf16_t)v;
}

__global__ __launch_bounds__(256)
void pack_b_kernel(const float* __restrict__ b_qkv, float* __restrict__ bias) {
    int idx = blockIdx.x * 256 + threadIdx.x;
    if (idx >= 3 * C_CH) return;
    int j = idx >> 9;
    int c = idx & 511;
    float v = b_qkv[3 * c + j];
    if (j < 2) v *= SCALE_QK;
    bias[idx] = v;
}

// ---------------------------------------------------------------------------
// Kernel 4: QKV GEMM.  Wave -> 16x64 tile; 8 waves/block cover all 512 cols.
// ---------------------------------------------------------------------------
__global__ __launch_bounds__(256)
void qkv_gemm_kernel(const bf16_t* __restrict__ xn,
                     const bf16_t* __restrict__ wt,     // [3][512][512]
                     const float* __restrict__ bias,    // [3][512]
                     bf16_t* __restrict__ qkv) {        // [3][4096][512]
    const int j    = blockIdx.y;
    const int m0   = blockIdx.x * 16;
    const int lane = threadIdx.x & 31;
    const int wv   = threadIdx.x >> 5;
    const int n0   = wv * 64;
    const int n    = lane & 15;
    const int hf   = lane >> 4;

    const bf16_t* W = wt + (size_t)j * C_CH * C_CH;
    bf16_t* out     = qkv + (size_t)j * T_TOK * C_CH;

    v8f acc[4];
#pragma unroll
    for (int nt = 0; nt < 4; ++nt) {
        float bv = bias[j * C_CH + n0 + nt * 16 + n];
#pragma unroll
        for (int r = 0; r < 8; ++r) acc[nt][r] = bv;
    }

    for (int k0 = 0; k0 < C_CH; k0 += 32) {
        v16bf Aa;
        const bf16_t* xr = &xn[(size_t)(m0 + n) * C_CH + k0];
#pragma unroll
        for (int i = 0; i < 8; ++i) {
            const bf16_t* p = &xr[a_kidx(i, hf)];
            Aa[2 * i]     = p[0];
            Aa[2 * i + 1] = p[1];
        }
#pragma unroll
        for (int nt = 0; nt < 4; ++nt) {
            v16bf Bb;
            const bf16_t* wr = W + (size_t)(n0 + nt * 16 + n) * C_CH + k0 + hf * 16;
#pragma unroll
            for (int i = 0; i < 8; ++i) {
                Bb[2 * i]     = wr[2 * i];
                Bb[2 * i + 1] = wr[2 * i + 1];
            }
            acc[nt] = __builtin_amdgcn_wmma_f32_16x16x32_bf16(
                false, Aa, false, Bb, (short)0, acc[nt], false, false);
        }
    }
#pragma unroll
    for (int nt = 0; nt < 4; ++nt)
#pragma unroll
        for (int r = 0; r < 8; ++r)
            out[(size_t)(m0 + r + hf * 8) * C_CH + n0 + nt * 16 + n] = (bf16_t)acc[nt][r];
}

// ---------------------------------------------------------------------------
// Kernel 5: flash attention with TRANSPOSED score tiles (S^T = K x Q^T) so the
// softmax key-reduction is in-lane + one shfl_xor(16) instead of 4-level trees.
// Block = 128 thr = 4 waves, one head; wave owns 16 queries; keys in 32-chunks.
// ---------------------------------------------------------------------------
__global__ __launch_bounds__(128)
void attn_kernel(const bf16_t* __restrict__ Qb,
                 const bf16_t* __restrict__ Kb,
                 const bf16_t* __restrict__ Vb,
                 float* __restrict__ out) {
    __shared__ bf16_t ldsK[32 * 64];      // [key][ch]
    __shared__ bf16_t ldsVT[64 * 32];     // [ch][key]
    __shared__ bf16_t ldsP[4][16 * 32];   // per-wave P staging [query][key]

    const int h    = blockIdx.y;
    const int tid  = threadIdx.x;
    const int lane = tid & 31;
    const int wv   = tid >> 5;
    const int q0   = blockIdx.x * 64 + wv * 16;
    const int n    = lane & 15;
    const int hf   = lane >> 4;

    // Preload Q^T as B-operands: element (kk,n) = Q[q0+n][h*64 + ks*32 + kk]
    v16bf Qt[2];
#pragma unroll
    for (int ks = 0; ks < 2; ++ks) {
        const bf16_t* p = &Qb[(size_t)(q0 + n) * C_CH + h * CH + ks * 32 + hf * 16];
#pragma unroll
        for (int i = 0; i < 8; ++i) {
            Qt[ks][2 * i]     = p[2 * i];
            Qt[ks][2 * i + 1] = p[2 * i + 1];
        }
    }

    v8f Ofrag[4];
#pragma unroll
    for (int nt = 0; nt < 4; ++nt)
#pragma unroll
        for (int r = 0; r < 8; ++r) Ofrag[nt][r] = 0.f;
    float mrun = -1e30f;   // running max for this lane's query (col lane&15)
    float lrun = 0.f;      // running sum

    for (int key0 = 0; key0 < T_TOK; key0 += 32) {
        __syncthreads();
        {   // cooperative stage: K row-major, V transposed
            int krow = tid >> 3;
            int ch8  = (tid & 7) << 3;
#pragma unroll
            for (int p = 0; p < 2; ++p) {
                int key = p * 16 + krow;
                size_t src = (size_t)(key0 + key) * C_CH + h * CH + ch8;
                uint4 kv = *(const uint4*)&Kb[src];
                *(uint4*)&ldsK[key * 64 + ch8] = kv;
                uint4 vv = *(const uint4*)&Vb[src];
                const bf16_t* ve = (const bf16_t*)&vv;
#pragma unroll
                for (int e = 0; e < 8; ++e) ldsVT[(ch8 + e) * 32 + key] = ve[e];
                if (key0 + 32 < T_TOK) {
                    __builtin_prefetch(&Kb[src + 32 * C_CH], 0, 0);
                    __builtin_prefetch(&Vb[src + 32 * C_CH], 0, 0);
                }
            }
        }
        __syncthreads();

        // S^T tiles: rows = keys, cols = queries.  A = K tile, B = Q^T.
        v8f ST[2];
#pragma unroll
        for (int kt = 0; kt < 2; ++kt) {
            v8f s;
#pragma unroll
            for (int r = 0; r < 8; ++r) s[r] = 0.f;
#pragma unroll
            for (int ks = 0; ks < 2; ++ks) {
                v16bf Ak;
                const bf16_t* kr = &ldsK[(kt * 16 + n) * 64 + ks * 32];
#pragma unroll
                for (int i = 0; i < 8; ++i) {
                    const bf16_t* p = &kr[a_kidx(i, hf)];
                    Ak[2 * i]     = p[0];
                    Ak[2 * i + 1] = p[1];
                }
                s = __builtin_amdgcn_wmma_f32_16x16x32_bf16(
                    false, Ak, false, Qt[ks], (short)0, s, false, false);
            }
            ST[kt] = s;
        }

        // Online softmax: this lane sees keys {kt*16 + r + 8*hf} for query n.
        float cm = -1e30f;
#pragma unroll
        for (int kt = 0; kt < 2; ++kt)
#pragma unroll
            for (int r = 0; r < 8; ++r) cm = fmaxf(cm, ST[kt][r]);
        cm = fmaxf(cm, __shfl_xor(cm, 16, 32));      // combine key halves
        float mnew  = fmaxf(mrun, cm);
        float alpha = __expf(mrun - mnew);
        mrun = mnew;

        float rs = 0.f;
        float pr[2][8];
#pragma unroll
        for (int kt = 0; kt < 2; ++kt)
#pragma unroll
            for (int r = 0; r < 8; ++r) {
                pr[kt][r] = __expf(ST[kt][r] - mnew);
                rs += pr[kt][r];
            }
        rs += __shfl_xor(rs, 16, 32);
        lrun = lrun * alpha + rs;

        // Stage P[query][key] (packed dword pairs: keys consecutive in r)
#pragma unroll
        for (int kt = 0; kt < 2; ++kt)
#pragma unroll
            for (int rp = 0; rp < 4; ++rp) {
                bf16x2 pk;
                pk[0] = (bf16_t)pr[kt][2 * rp];
                pk[1] = (bf16_t)pr[kt][2 * rp + 1];
                *(bf16x2*)&ldsP[wv][n * 32 + kt * 16 + hf * 8 + 2 * rp] = pk;
            }

        // Broadcast alpha to O-layout rows (row r+8*hf holds query r+8*hf)
        float ar[8];
#pragma unroll
        for (int r = 0; r < 8; ++r) ar[r] = __shfl(alpha, r + 8 * hf, 32);
#pragma unroll
        for (int nt = 0; nt < 4; ++nt)
#pragma unroll
            for (int r = 0; r < 8; ++r) Ofrag[nt][r] *= ar[r];

        asm volatile("s_wait_dscnt 0" ::: "memory");  // cross-lane LDS RAW

        // Read P back in A layout (M=query, K=key)
        v16bf Pa;
#pragma unroll
        for (int i = 0; i < 8; ++i) {
            const bf16_t* pp = &ldsP[wv][n * 32 + a_kidx(i, hf)];
            Pa[2 * i]     = pp[0];
            Pa[2 * i + 1] = pp[1];
        }
        // O += P x V
#pragma unroll
        for (int nt = 0; nt < 4; ++nt) {
            v16bf Bv;
            const bf16_t* vr = &ldsVT[(nt * 16 + n) * 32 + hf * 16];
#pragma unroll
            for (int i = 0; i < 8; ++i) {
                Bv[2 * i]     = vr[2 * i];
                Bv[2 * i + 1] = vr[2 * i + 1];
            }
            Ofrag[nt] = __builtin_amdgcn_wmma_f32_16x16x32_bf16(
                false, Pa, false, Bv, (short)0, Ofrag[nt], false, false);
        }
    }

    // Finalize: fetch row sums from the lane owning each query, divide, store.
#pragma unroll
    for (int r = 0; r < 8; ++r) {
        float lr  = __shfl(lrun, r + 8 * hf, 32);
        float inv = 1.f / lr;
        int row = q0 + r + 8 * hf;
#pragma unroll
        for (int nt = 0; nt < 4; ++nt)
            out[(size_t)row * C_CH + h * CH + nt * 16 + n] = Ofrag[nt][r] * inv;
    }
}

// ---------------------------------------------------------------------------
extern "C" void kernel_launch(void* const* d_in, const int* in_sizes, int n_in,
                              void* d_out, int out_size, void* d_ws, size_t ws_size,
                              hipStream_t stream) {
    const float* x     = (const float*)d_in[0];
    const float* gs    = (const float*)d_in[1];
    const float* gb    = (const float*)d_in[2];
    const float* w_qkv = (const float*)d_in[3];
    const float* b_qkv = (const float*)d_in[4];
    float* out         = (float*)d_out;

    char* ws = (char*)d_ws;
    float*  acoef = (float*)(ws + 0);                        // 512 f32
    float*  bcoef = (float*)(ws + 2048);                     // 512 f32
    float*  bias  = (float*)(ws + 4096);                     // 3*512 f32
    bf16_t* wt    = (bf16_t*)(ws + 10240);                   // 3*512*512 bf16
    bf16_t* xn    = (bf16_t*)(ws + 10240 + (size_t)3 * C_CH * C_CH * 2);   // T*C bf16
    bf16_t* qkv   = (bf16_t*)(ws + 10240 + (size_t)3 * C_CH * C_CH * 2
                                        + (size_t)T_TOK * C_CH * 2);       // 3*T*C bf16

    gn_stats_kernel<<<NG, 256, 0, stream>>>(x, gs, gb, acoef, bcoef);
    xn_norm_kernel<<<(T_TOK * C_CH / 4) / 256, 256, 0, stream>>>(x, acoef, bcoef, xn);
    pack_w_kernel<<<(3 * C_CH * C_CH) / 256, 256, 0, stream>>>(w_qkv, wt);
    pack_b_kernel<<<6, 256, 0, stream>>>(b_qkv, bias);
    qkv_gemm_kernel<<<dim3(T_TOK / 16, 3), 256, 0, stream>>>(xn, wt, bias, qkv);
    attn_kernel<<<dim3(T_TOK / 64, NH), 128, 0, stream>>>(
        qkv, qkv + (size_t)T_TOK * C_CH, qkv + (size_t)2 * T_TOK * C_CH, out);
}
